// RNNLayer_80229989089777
// MI455X (gfx1250) — matrix operand order repaired and checked by reading, not compiled, not run on hardware
//
#include <hip/hip_runtime.h>
#include <math.h>

#define B_DIM  4096
#define H_DIM  1024
#define NI_DIM 4096
#define NQ_DIM 2048

typedef _Float16 h8_t  __attribute__((ext_vector_type(8)));
typedef _Float16 v16h  __attribute__((ext_vector_type(16)));
typedef float    v8f   __attribute__((ext_vector_type(8)));
typedef float    f4_t  __attribute__((ext_vector_type(4)));

union V16U { v16h v; h8_t h[2]; };

// ---------------------------------------------------------------------------
// Kernel A: hidden = tanh(state @ W_t.T + inputX @ W_x.T + b_t + b_x)
// Fused K = 1024 + 4096 GEMM. f32 tiles staged as f16 in LDS (double buffered,
// K=64 per stage = 2 WMMA sub-steps), v_wmma_f32_16x16x32_f16.
// Block = 256 threads = 8 waves (4 M x 2 N). C tile = 64 x 128.
// ---------------------------------------------------------------------------

#define KTILE   64                    // K staged per barrier
#define LDS_LD  72                    // 64 halves + 8 pad (144B rows, 16B aligned)
#define NSTAGE1 (H_DIM  / KTILE)      // 16
#define NSTAGE2 (NI_DIM / KTILE)      // 64

// Stage one 64xKTILE A tile + 128xKTILE B tile, converting f32 -> f16.
// Each thread: A -> 16 contiguous floats (64B, 4x b128), B -> 2 x 16 floats.
__device__ __forceinline__ void stage_tile(
    const float* __restrict__ A, int lda,
    const float* __restrict__ W, int ldw,
    int row0, int col0, int kk, int tid,
    _Float16* __restrict__ As, _Float16* __restrict__ Bs)
{
    const int r = tid >> 2;           // 0..63
    const int c = (tid & 3) * 16;     // 0,16,32,48
    {
        const float* p = A + (size_t)(row0 + r) * lda + kk + c;
        h8_t h0, h1;
        #pragma unroll
        for (int j = 0; j < 2; ++j) {
            f4_t x0 = *(const f4_t*)(p + j * 8);
            f4_t x1 = *(const f4_t*)(p + j * 8 + 4);
            h8_t h;
            h[0]=(_Float16)x0[0]; h[1]=(_Float16)x0[1]; h[2]=(_Float16)x0[2]; h[3]=(_Float16)x0[3];
            h[4]=(_Float16)x1[0]; h[5]=(_Float16)x1[1]; h[6]=(_Float16)x1[2]; h[7]=(_Float16)x1[3];
            if (j == 0) h0 = h; else h1 = h;
        }
        *(h8_t*)&As[r * LDS_LD + c]     = h0;
        *(h8_t*)&As[r * LDS_LD + c + 8] = h1;
    }
    #pragma unroll
    for (int it = 0; it < 2; ++it) {
        const int rr = r + it * 64;   // 0..127 weight rows (= output columns)
        const float* p = W + (size_t)(col0 + rr) * ldw + kk + c;
        h8_t h0, h1;
        #pragma unroll
        for (int j = 0; j < 2; ++j) {
            f4_t x0 = *(const f4_t*)(p + j * 8);
            f4_t x1 = *(const f4_t*)(p + j * 8 + 4);
            h8_t h;
            h[0]=(_Float16)x0[0]; h[1]=(_Float16)x0[1]; h[2]=(_Float16)x0[2]; h[3]=(_Float16)x0[3];
            h[4]=(_Float16)x1[0]; h[5]=(_Float16)x1[1]; h[6]=(_Float16)x1[2]; h[7]=(_Float16)x1[3];
            if (j == 0) h0 = h; else h1 = h;
        }
        *(h8_t*)&Bs[rr * LDS_LD + c]     = h0;
        *(h8_t*)&Bs[rr * LDS_LD + c + 8] = h1;
    }
}

// One staged tile: 2 K-substeps x 4 N-tiles of WMMA.
// All fragments for a substep are loaded before any WMMA issues so a single
// s_wait_dscnt covers them and the 4 WMMAs go back-to-back.
__device__ __forceinline__ void compute_tile(
    const _Float16* __restrict__ As, const _Float16* __restrict__ Bs,
    int aRow, int wn, int hi, int nl, v8f acc[4])
{
    #pragma unroll
    for (int ks = 0; ks < KTILE; ks += 32) {
        V16U a;
        a.h[0] = *(const h8_t*)&As[aRow * LDS_LD + ks + hi * 8];
        a.h[1] = *(const h8_t*)&As[aRow * LDS_LD + ks + 16 + hi * 8];
        V16U b[4];
        #pragma unroll
        for (int nt = 0; nt < 4; ++nt) {
            const int bcol = wn * 64 + nt * 16 + nl;
            b[nt].h[0] = *(const h8_t*)&Bs[bcol * LDS_LD + ks + hi * 8];
            b[nt].h[1] = *(const h8_t*)&Bs[bcol * LDS_LD + ks + 16 + hi * 8];
        }
        #pragma unroll
        for (int nt = 0; nt < 4; ++nt)
            acc[nt] = __builtin_amdgcn_wmma_f32_16x16x32_f16(
                false, a.v, false, b[nt].v, (short)0, acc[nt], false, false);
    }
}

__global__ __launch_bounds__(256) void hidden_gemm_kernel(
    const float* __restrict__ state, const float* __restrict__ inputX,
    const float* __restrict__ W_t,   const float* __restrict__ b_t,
    const float* __restrict__ W_x,   const float* __restrict__ b_x,
    float* __restrict__ hidden_out)
{
    __shared__ __align__(16) _Float16 As[2][64  * LDS_LD];
    __shared__ __align__(16) _Float16 Bs[2][128 * LDS_LD];

    const int tid  = threadIdx.x;
    const int lane = tid & 31;
    const int w    = tid >> 5;
    const int wm   = w & 3;
    const int wn   = w >> 2;
    const int hi   = lane >> 4;
    const int nl   = lane & 15;

    const int row0 = blockIdx.x * 64;
    const int col0 = blockIdx.y * 128;
    const int aRow = wm * 16 + nl;

    v8f acc[4];
    #pragma unroll
    for (int i = 0; i < 4; ++i) acc[i] = (v8f)(0.f);

    int cur = 0;
    stage_tile(state, H_DIM, W_t, H_DIM, row0, col0, 0, tid, As[0], Bs[0]);
    __syncthreads();

    // Segment 1: K over state / W_t  (lda = ldw = H_DIM)
    for (int s = 0; s < NSTAGE1; ++s) {
        if (s + 1 < NSTAGE1)
            stage_tile(state, H_DIM, W_t, H_DIM, row0, col0, (s + 1) * KTILE,
                       tid, As[cur ^ 1], Bs[cur ^ 1]);
        else
            stage_tile(inputX, NI_DIM, W_x, NI_DIM, row0, col0, 0,
                       tid, As[cur ^ 1], Bs[cur ^ 1]);
        if (s + 2 < NSTAGE1) {
            const int kk = (s + 2) * KTILE;
            __builtin_prefetch(state + (size_t)(row0 + (tid >> 2)) * H_DIM + kk, 0, 1);
            __builtin_prefetch(W_t   + (size_t)(col0 + (tid >> 2)) * H_DIM + kk, 0, 1);
        }
        compute_tile(As[cur], Bs[cur], aRow, wn, hi, nl, acc);
        __syncthreads();
        cur ^= 1;
    }

    // Segment 2: K over inputX / W_x  (lda = ldw = NI_DIM)
    for (int s = 0; s < NSTAGE2; ++s) {
        if (s + 1 < NSTAGE2)
            stage_tile(inputX, NI_DIM, W_x, NI_DIM, row0, col0, (s + 1) * KTILE,
                       tid, As[cur ^ 1], Bs[cur ^ 1]);
        if (s + 2 < NSTAGE2) {
            const int kk = (s + 2) * KTILE;
            __builtin_prefetch(inputX + (size_t)(row0 + (tid >> 2)) * NI_DIM + kk, 0, 1);
            __builtin_prefetch(W_x    + (size_t)(col0 + (tid >> 2)) * NI_DIM + kk, 0, 1);
        }
        compute_tile(As[cur], Bs[cur], aRow, wn, hi, nl, acc);
        __syncthreads();
        cur ^= 1;
    }

    // Epilogue: bias + tanh, write hidden (row-major B_DIM x H_DIM)
    const int mBase = row0 + wm * 16 + hi * 8;
    #pragma unroll
    for (int nt = 0; nt < 4; ++nt) {
        const int n = col0 + wn * 64 + nt * 16 + nl;
        const float bsum = b_t[n] + b_x[n];
        #pragma unroll
        for (int r = 0; r < 8; ++r) {
            const float v = tanhf(acc[nt][r] + bsum);
            hidden_out[(size_t)(mBase + r) * H_DIM + n] = v;
        }
    }
}

// ---------------------------------------------------------------------------
// Kernel B: one-hot gather + dot(hidden_row, W_y[q]) + sigmoid + BCE term.
// Exact rewrite: sum_j sigmoid(z_j)*onehot_j == sigmoid(z_q).
// ---------------------------------------------------------------------------
__global__ __launch_bounds__(256) void pred_bce_kernel(
    const float* __restrict__ inputY, const float* __restrict__ truth,
    const float* __restrict__ W_y,    const float* __restrict__ b_y,
    const float* __restrict__ hidden, float* __restrict__ pred_out,
    float* __restrict__ bce_ws)
{
    __shared__ int   qsh;
    __shared__ float red[256];

    const int b   = blockIdx.x;
    const int tid = threadIdx.x;
    if (tid == 0) qsh = 0;
    __syncthreads();

    const float* yrow = inputY + (size_t)b * NQ_DIM;
    for (int i = tid; i < NQ_DIM; i += 256)
        if (yrow[i] > 0.5f) qsh = i;       // exactly one hit per row
    __syncthreads();
    const int q = qsh;

    const float* hrow = hidden + (size_t)b * H_DIM;
    const float* wrow = W_y    + (size_t)q * H_DIM;
    float s = 0.f;
    for (int i = tid; i < H_DIM; i += 256)
        s = fmaf(hrow[i], wrow[i], s);
    red[tid] = s;
    __syncthreads();
    #pragma unroll
    for (int off = 128; off > 0; off >>= 1) {
        if (tid < off) red[tid] += red[tid + off];
        __syncthreads();
    }
    if (tid == 0) {
        const float z = red[0] + b_y[q];
        const float p = 1.f / (1.f + expf(-z));
        pred_out[b] = p;
        const float t   = truth[b];
        const float lp  = fmaxf(logf(p),    -100.f);
        const float l1p = fmaxf(log1pf(-p), -100.f);
        bce_ws[b] = -(t * lp + (1.f - t) * l1p);
    }
}

// ---------------------------------------------------------------------------
// Kernel C: deterministic tree reduction of 4096 BCE terms -> err scalar
// ---------------------------------------------------------------------------
__global__ __launch_bounds__(1024) void bce_reduce_kernel(
    const float* __restrict__ bce_ws, float* __restrict__ err_out)
{
    __shared__ float red[1024];
    const int tid = threadIdx.x;
    red[tid] = bce_ws[tid] + bce_ws[tid + 1024] + bce_ws[tid + 2048] + bce_ws[tid + 3072];
    __syncthreads();
    #pragma unroll
    for (int off = 512; off > 0; off >>= 1) {
        if (tid < off) red[tid] += red[tid + off];
        __syncthreads();
    }
    if (tid == 0) err_out[0] = red[0];
}

// ---------------------------------------------------------------------------
// d_in order: state, inputX, inputY, truth, W_t, b_t, W_x, b_x, W_y, b_y
// d_out layout: pred[4096] | err[1] | hidden[4096*1024]
// ---------------------------------------------------------------------------
extern "C" void kernel_launch(void* const* d_in, const int* in_sizes, int n_in,
                              void* d_out, int out_size, void* d_ws, size_t ws_size,
                              hipStream_t stream) {
    const float* state  = (const float*)d_in[0];
    const float* inputX = (const float*)d_in[1];
    const float* inputY = (const float*)d_in[2];
    const float* truth  = (const float*)d_in[3];
    const float* W_t    = (const float*)d_in[4];
    const float* b_t    = (const float*)d_in[5];
    const float* W_x    = (const float*)d_in[6];
    const float* b_x    = (const float*)d_in[7];
    const float* W_y    = (const float*)d_in[8];
    const float* b_y    = (const float*)d_in[9];

    float* pred   = (float*)d_out;                  // [0 .. 4095]
    float* err    = (float*)d_out + B_DIM;          // [4096]
    float* hidden = (float*)d_out + B_DIM + 1;      // [4097 ..]
    float* bce_ws = (float*)d_ws;                   // 4096 floats

    dim3 gridA(B_DIM / 64, H_DIM / 128);
    hidden_gemm_kernel<<<gridA, 256, 0, stream>>>(state, inputX, W_t, b_t, W_x, b_x, hidden);

    pred_bce_kernel<<<dim3(B_DIM), 256, 0, stream>>>(inputY, truth, W_y, b_y, hidden, pred, bce_ws);

    bce_reduce_kernel<<<dim3(1), 1024, 0, stream>>>(bce_ws, err);
}
